// ManifoldConstrainedHyperConnections_13168369729906
// MI455X (gfx1250) — matrix operand order repaired and checked by reading, not compiled, not run on hardware
//
#include <hip/hip_runtime.h>
#include <hip/hip_bf16.h>
#include <math.h>

typedef __attribute__((ext_vector_type(16))) __bf16 v16bf;
typedef __attribute__((ext_vector_type(8)))  float  v8f;
typedef __attribute__((ext_vector_type(4)))  float  v4f;

#define CH        2048            // n*C = 4*512
#define TOKBLK    16              // tokens per block (one WMMA M-tile)
#define LSTRIDE   2056            // padded LDS row stride (floats): 2-way-max bank conflicts
#define RMS_EPS   1.1920929e-07f
#define SK_EPS    1e-8f

// ---------------------------------------------------------------------------
// Prologue: pack [w_pre(4); w_post(4); w_res(16); zeros(8)] x 2048 into
// bf16 hi/lo pairs for bf16x3 fp32-emulated WMMA. W stays hot in L2 (RT hint).
// ---------------------------------------------------------------------------
__global__ void mchc_convw(const float* __restrict__ wpre,
                           const float* __restrict__ wpost,
                           const float* __restrict__ wres,
                           __bf16* __restrict__ Whi,
                           __bf16* __restrict__ Wlo) {
    int idx = blockIdx.x * 256 + threadIdx.x;
    if (idx >= 32 * CH) return;
    int r = idx / CH, k = idx % CH;
    float v = 0.0f;
    if (r < 4)       v = wpre[r * CH + k];
    else if (r < 8)  v = wpost[(r - 4) * CH + k];
    else if (r < 24) v = wres[(r - 8) * CH + k];
    __bf16 h = (__bf16)v;
    Whi[idx] = h;
    Wlo[idx] = (__bf16)(v - (float)h);
}

// ---------------------------------------------------------------------------
// Fused main kernel: 1 block = 16 tokens, 4 waves.
// ---------------------------------------------------------------------------
__global__ __launch_bounds__(128)
void mchc_main(const float* __restrict__ x,
               const float* __restrict__ b_pre,
               const float* __restrict__ b_post,
               const float* __restrict__ b_res,
               const float* __restrict__ a_pre,
               const float* __restrict__ a_post,
               const float* __restrict__ a_res,
               const __bf16* __restrict__ Whi,
               const __bf16* __restrict__ Wlo,
               float* __restrict__ out) {
    extern __shared__ float smem[];
    float* xs   = smem;                       // [16][LSTRIDE] f32 x tile
    float* part = xs + TOKBLK * LSTRIDE;      // [2][16][32] K-half partials
    float* ssq  = part + 2 * 16 * 32;         // [16] sum of squares
    float* coef = ssq + 16;                   // [16][16] combined mix coeffs

    const int  tid  = threadIdx.x;
    const int  lane = tid & 31;
    const int  wave = tid >> 5;
    const long tb   = (long)blockIdx.x * TOKBLK;

    // ---- Stage 1: global -> LDS tile load (NT: streamed, no reuse in L2)
    //      + per-token sum(x^2) computed in-flight ------------------------
    for (int tt = 0; tt < 4; ++tt) {
        const int t = wave * 4 + tt;
        const v4f* xp = (const v4f*)(x + (tb + t) * (long)CH);
        float ss = 0.0f;
        #pragma unroll 4
        for (int i = lane; i < CH / 4; i += 32) {
            v4f v = __builtin_nontemporal_load(&xp[i]);
            *(v4f*)&xs[t * LSTRIDE + i * 4] = v;
            ss += v[0]*v[0] + v[1]*v[1] + v[2]*v[2] + v[3]*v[3];
        }
        for (int off = 16; off > 0; off >>= 1)
            ss += __shfl_xor(ss, off, 32);
        if (lane == 0) ssq[t] = ss;
    }
    __syncthreads();

    // ---- Stage 2: WMMA projections (bf16x3 fp32 emulation) ----------------
    {
        const int ct   = wave & 1;            // column tile (W rows 0-15 / 16-31)
        const int kh   = wave >> 1;           // K half
        const int m    = lane & 15;           // token row of this lane (A layout)
        const int koff = (lane >> 4) * 8;     // A-layout K offset per half-wave
        const int nrow = ct * 16 + (lane & 15);
        const int kq   = (lane >> 4) * 16;    // B-layout K offset per half-wave
        const float sm = 1.0f / sqrtf(ssq[m] * (1.0f / CH) + RMS_EPS);

        v8f c = {};
        const int kbeg = kh * (CH / 2);
        const int kend = kbeg + (CH / 2);
        for (int kb = kbeg; kb < kend; kb += 32) {
            // A fragment: 16 bf16/lane, K order = {kq..kq+15}
            const float* rowp = &xs[m * LSTRIDE + kb];
            v4f p[4];
            p[0] = *(const v4f*)(rowp + kq);
            p[1] = *(const v4f*)(rowp + kq + 4);
            p[2] = *(const v4f*)(rowp + kq + 8);
            p[3] = *(const v4f*)(rowp + kq + 12);
            v16bf ahi, alo;
            #pragma unroll
            for (int q = 0; q < 4; ++q)
                #pragma unroll
                for (int e = 0; e < 4; ++e) {
                    int j = q * 4 + e;
                    float v = p[q][e] * sm;          // apply RMSNorm scale
                    __bf16 h = (__bf16)v;
                    ahi[j] = h;
                    alo[j] = (__bf16)(v - (float)h);
                }
            // B fragments: contiguous 16 bf16 of weight row `nrow` (RT, L2-hot)
            const v16bf bhi = *(const v16bf*)(Whi + (long)nrow * CH + kb + kq);
            const v16bf blo = *(const v16bf*)(Wlo + (long)nrow * CH + kb + kq);

            c = __builtin_amdgcn_wmma_f32_16x16x32_bf16(false, ahi, false, bhi,
                                                        (short)0, c, false, false);
            c = __builtin_amdgcn_wmma_f32_16x16x32_bf16(false, ahi, false, blo,
                                                        (short)0, c, false, false);
            c = __builtin_amdgcn_wmma_f32_16x16x32_bf16(false, alo, false, bhi,
                                                        (short)0, c, false, false);
        }
        // Write partials: C layout: VGPR r -> M = r + 8*(lane>>4), N = lane&15
        const int M0 = (lane >> 4) * 8;
        const int N  = lane & 15;
        #pragma unroll
        for (int r = 0; r < 8; ++r)
            part[kh * 512 + (M0 + r) * 32 + ct * 16 + N] = c[r];
    }
    __syncthreads();

    // ---- Stage 3: gates + Sinkhorn-Knopp (1 token per lane, 16 lanes) -----
    if (tid < TOKBLK) {
        const int t = tid;
        float L[24];
        #pragma unroll
        for (int j = 0; j < 24; ++j)
            L[j] = part[t * 32 + j] + part[512 + t * 32 + j];

        const float ap = a_pre[0], apo = a_post[0], ar = a_res[0];
        float Hpre[4], Hpost[4];
        #pragma unroll
        for (int i = 0; i < 4; ++i) {
            float r1 = ap  * tanhf(L[i])     + b_pre[i];
            float r2 = apo * tanhf(L[4 + i]) + b_post[i];
            Hpre[i]  = 1.0f / (1.0f + expf(-r1));
            Hpost[i] = 2.0f / (1.0f + expf(-r2));
        }
        float P[16];
        #pragma unroll
        for (int j = 0; j < 16; ++j)
            P[j] = expf(ar * tanhf(L[8 + j]) + b_res[j]);
        for (int it = 0; it < 20; ++it) {
            #pragma unroll
            for (int n = 0; n < 4; ++n) {
                float inv = 1.0f / (P[n*4]+P[n*4+1]+P[n*4+2]+P[n*4+3] + SK_EPS);
                #pragma unroll
                for (int mm = 0; mm < 4; ++mm) P[n*4+mm] *= inv;
            }
            #pragma unroll
            for (int mm = 0; mm < 4; ++mm) {
                float inv = 1.0f / (P[mm]+P[4+mm]+P[8+mm]+P[12+mm] + SK_EPS);
                #pragma unroll
                for (int n = 0; n < 4; ++n) P[n*4+mm] *= inv;
            }
        }
        // out[n] = sum_m (Hres[n][m] + Hpost[n]*Hpre[m]) * x[m]
        #pragma unroll
        for (int n = 0; n < 4; ++n)
            #pragma unroll
            for (int mm = 0; mm < 4; ++mm)
                coef[t * 16 + n * 4 + mm] = P[n*4+mm] + Hpost[n] * Hpre[mm];
    }
    __syncthreads();

    // ---- Stage 4: recombine from LDS tile, coalesced NT float4 stores -----
    for (int j = tid; j < TOKBLK * 4 * 128; j += 128) {
        const int t  = j >> 9;
        const int n  = (j >> 7) & 3;
        const int cg = j & 127;               // float4 group within C=512
        const float* cf = &coef[t * 16 + n * 4];
        v4f acc = {};
        #pragma unroll
        for (int mm = 0; mm < 4; ++mm) {
            v4f xv = *(const v4f*)&xs[t * LSTRIDE + mm * 512 + cg * 4];
            acc += cf[mm] * xv;
        }
        __builtin_nontemporal_store(acc,
            (v4f*)(out + (((tb + t) * 4 + n) * 512 + cg * 4)));
    }
}

// ---------------------------------------------------------------------------
extern "C" void kernel_launch(void* const* d_in, const int* in_sizes, int n_in,
                              void* d_out, int out_size, void* d_ws, size_t ws_size,
                              hipStream_t stream) {
    const float* x      = (const float*)d_in[0];
    const float* w_pre  = (const float*)d_in[1];
    const float* w_post = (const float*)d_in[2];
    const float* w_res  = (const float*)d_in[3];
    const float* b_pre  = (const float*)d_in[4];
    const float* b_post = (const float*)d_in[5];
    const float* b_res  = (const float*)d_in[6];
    const float* a_pre  = (const float*)d_in[7];
    const float* a_post = (const float*)d_in[8];
    const float* a_res  = (const float*)d_in[9];
    float* out = (float*)d_out;

    __bf16* Whi = (__bf16*)d_ws;              // [32][2048] bf16 hi (128 KB)
    __bf16* Wlo = Whi + 32 * CH;              // [32][2048] bf16 lo (128 KB)

    const int tokens = in_sizes[0] / CH;      // B*L = 16384

    hipLaunchKernelGGL(mchc_convw, dim3((32 * CH + 255) / 256), dim3(256), 0,
                       stream, w_pre, w_post, w_res, Whi, Wlo);

    const size_t lds = (TOKBLK * LSTRIDE + 2 * 16 * 32 + 16 + 256) * sizeof(float);
    hipLaunchKernelGGL(mchc_main, dim3(tokens / TOKBLK), dim3(128), lds, stream,
                       x, b_pre, b_post, b_res, a_pre, a_post, a_res,
                       Whi, Wlo, out);
}
